// TransformerBlock_83829171683956
// MI455X (gfx1250) — matrix operand order repaired and checked by reading, not compiled, not run on hardware
//
#include <hip/hip_runtime.h>
#include <hip/hip_bf16.h>
#include <stdint.h>

// ---------------------------------------------------------------------------
// CDNA5 (gfx1250) bf16 WMMA transformer block + 8-qubit statevector layer.
// B=4, S=1024, D=1024, H=16, dk=64, F=4096, tokens M=4096.
// GEMMs: TDM (tensor_load_to_lds) double-buffered LDS staging + WMMA.
// ---------------------------------------------------------------------------

typedef __attribute__((ext_vector_type(16))) __bf16 v16bf;
typedef __attribute__((ext_vector_type(8)))  float  v8f;
typedef __attribute__((ext_vector_type(4)))  unsigned int u32x4;
typedef __attribute__((ext_vector_type(8)))  int  i32x8;
typedef __attribute__((ext_vector_type(4)))  int  i32x4;

union Frag {
  v16bf v;
  uint4 q[2];
};

__device__ __forceinline__ v8f wmma_bf16(const Frag& a, const Frag& b, v8f c) {
  return __builtin_amdgcn_wmma_f32_16x16x32_bf16(false, a.v, false, b.v,
                                                 (short)0, c, false, false);
}

// A fragment: 16(M) x 32(K) bf16 from row-major [.,K] source, leading dim lda.
// ISA layout: lane holds row m=lane&15; hi=lane>>4; element e: K = (e<8) ?
// k0+8*hi+e : k0+16+8*hi+(e-8)  -> two contiguous 16B runs.
__device__ __forceinline__ void load_a(Frag& f, const __bf16* A, int lda,
                                       int m0, int k0, int lane) {
  const int m  = m0 + (lane & 15);
  const int hi = lane >> 4;
  const __bf16* p = A + (size_t)m * lda + k0 + 8 * hi;
  f.q[0] = *(const uint4*)p;
  f.q[1] = *(const uint4*)(p + 16);
}

// B fragment: 32(K) x 16(N) from row-major weight [N,K], leading dim ldb.
// ISA layout: lane holds column n=lane&15; element e: K = (e<8) ?
// k0+8*hi+e : k0+16+8*hi+(e-8)  -> two contiguous 16B runs.
__device__ __forceinline__ void load_b(Frag& f, const __bf16* B, int ldb,
                                       int n0, int k0, int lane) {
  const int n  = n0 + (lane & 15);
  const int hi = lane >> 4;
  const __bf16* p = B + (size_t)n * ldb + k0 + 8 * hi;
  f.q[0] = *(const uint4*)p;
  f.q[1] = *(const uint4*)(p + 16);
}

// ---------------------------------------------------------------------------
// Tensor Data Mover: DMA a [rows x 32] bf16 tile (row pitch = pitch elements)
// from global memory into LDS (contiguous rows of 64B). 2D tile -> groups 2/3
// are zero. D# per CDNA5 ISA. 6-arg builtin (clang-23).
// ---------------------------------------------------------------------------
__device__ __forceinline__ void tdm_load_tile(unsigned int lds_addr,
                                              const __bf16* gptr,
                                              int pitch, int tile_k, int tile_rows) {
  const unsigned long long ga = (unsigned long long)(uintptr_t)gptr;
  u32x4 g0;
  g0.x = 1u;                                    // count=1, user descriptor
  g0.y = lds_addr;                              // LDS byte address
  g0.z = (unsigned int)ga;                      // global_addr[31:0]
  g0.w = (unsigned int)(ga >> 32) | (2u << 30); // global_addr[56:32] | type=2
  i32x8 g1;
  g1[0] = 1 << 16;                                          // data_size=1 (2B)
  g1[1] = (tile_k & 0xffff) << 16;                          // tensor_dim0 lo
  g1[2] = (tile_k >> 16) | ((tile_rows & 0xffff) << 16);    // dim0 hi | tensor_dim1 lo
  g1[3] = ((unsigned)tile_rows >> 16) | (tile_k << 16);     // dim1 hi | tile_dim0
  g1[4] = tile_rows & 0xffff;                               // tile_dim1 (tile_dim2=0)
  g1[5] = pitch;                                            // tensor_dim0_stride lo32
  g1[6] = 0;
  g1[7] = 0;
  const i32x4 z4 = {0, 0, 0, 0};
  const i32x8 z8 = {0, 0, 0, 0, 0, 0, 0, 0};
  __builtin_amdgcn_tensor_load_to_lds(g0, g1, z4, z4, z8, 0);
}

__device__ __forceinline__ unsigned int lds_off(const void* p) {
  return (unsigned int)(uintptr_t)p;   // low 32 bits of LDS aperture address
}

// ---------------------------------------------------------------------------
// fp32 -> bf16 conversion, 8 elements / thread (b128 in, b128 out)
// ---------------------------------------------------------------------------
__global__ void cvt_bf16_kernel(const float* __restrict__ in,
                                __bf16* __restrict__ out, long n8) {
  long i = (long)blockIdx.x * blockDim.x + threadIdx.x;
  const long stride = (long)gridDim.x * blockDim.x;
  for (; i < n8; i += stride) {
    const float4* p = (const float4*)(in + i * 8);
    const float4 f0 = p[0], f1 = p[1];
    union { __bf16 h[8]; uint4 u; } o;
    o.h[0] = (__bf16)f0.x; o.h[1] = (__bf16)f0.y;
    o.h[2] = (__bf16)f0.z; o.h[3] = (__bf16)f0.w;
    o.h[4] = (__bf16)f1.x; o.h[5] = (__bf16)f1.y;
    o.h[6] = (__bf16)f1.z; o.h[7] = (__bf16)f1.w;
    *(uint4*)(out + i * 8) = o.u;
  }
}

// ---------------------------------------------------------------------------
// GEMM:  C[M,N] = A[M,K] (bf16) * W[N,K]^T (bf16), TDM-staged, double-buffered.
// block = 256 threads (8 waves), tile 64(M) x 256(N); wave tile 32x64.
// ---------------------------------------------------------------------------
enum { EPI_F32 = 0, EPI_RELU_BF16 = 1, EPI_QKV = 2, EPI_VT = 3 };

template <int EPI>
__global__ __launch_bounds__(256)
void gemm_bf16_kernel(const __bf16* __restrict__ A, const __bf16* __restrict__ W,
                      int K, int lda, int ldb,
                      const float* __restrict__ bias,
                      const float* __restrict__ res,
                      float* __restrict__ Cf, __bf16* __restrict__ Cb, int ldc) {
  __shared__ __align__(16) __bf16 shA[2][64 * 32];    //  4KB x2
  __shared__ __align__(16) __bf16 shB[2][256 * 32];   // 16KB x2
  const int lane = threadIdx.x & 31;
  const int w    = threadIdx.x >> 5;
  const int wm   = w >> 2;           // 0..1
  const int wn   = w & 3;            // 0..3
  const int mblk = blockIdx.y * 64;
  const int nblk = blockIdx.x * 256;
  const int nsteps = K >> 5;
  const bool issuer = (threadIdx.x < 32);   // wave 0 drives the TDM

  if (issuer) {
    tdm_load_tile(lds_off(&shA[0][0]), A + (size_t)mblk * lda, lda, 32, 64);
    tdm_load_tile(lds_off(&shB[0][0]), W + (size_t)nblk * ldb, ldb, 32, 256);
  }

  v8f acc[2][4] = {};
  for (int it = 0; it < nsteps; ++it) {
    const int buf = it & 1;
    if (issuer) {
      if (it + 1 < nsteps) {
        const int k0 = (it + 1) << 5;
        tdm_load_tile(lds_off(&shA[buf ^ 1][0]), A + (size_t)mblk * lda + k0, lda, 32, 64);
        tdm_load_tile(lds_off(&shB[buf ^ 1][0]), W + (size_t)nblk * ldb + k0, ldb, 32, 256);
        __builtin_amdgcn_s_wait_tensorcnt(2);   // current pair has landed
      } else {
        __builtin_amdgcn_s_wait_tensorcnt(0);
      }
    }
    __syncthreads();                            // tile `buf` visible to all waves

    Frag a0, a1;
    load_a(a0, &shA[buf][0], 32, wm * 32,      0, lane);
    load_a(a1, &shA[buf][0], 32, wm * 32 + 16, 0, lane);
#pragma unroll
    for (int j = 0; j < 4; j++) {
      Frag b;
      load_b(b, &shB[buf][0], 32, wn * 64 + 16 * j, 0, lane);
      acc[0][j] = wmma_bf16(a0, b, acc[0][j]);
      acc[1][j] = wmma_bf16(a1, b, acc[1][j]);
    }
    __syncthreads();                            // done reading before overwrite
  }

  // C layout: VGPR r, lanes 0-15 -> row r, lanes 16-31 -> row r+8; col = lane&15.
  const int m0 = mblk + wm * 32;
  const int n0 = nblk + wn * 64;
  const int nlane  = lane & 15;
  const int rowadd = (lane < 16) ? 0 : 8;
#pragma unroll
  for (int i = 0; i < 2; i++) {
#pragma unroll
    for (int j = 0; j < 4; j++) {
#pragma unroll
      for (int r = 0; r < 8; r++) {
        const int m = m0 + 16 * i + rowadd + r;
        const int n = n0 + 16 * j + nlane;
        float v = acc[i][j][r];
        if (EPI == EPI_F32) {
          if (bias) v += bias[n];
          if (res)  v += res[(size_t)m * ldc + n];
          Cf[(size_t)m * ldc + n] = v;
        } else if (EPI == EPI_RELU_BF16) {
          v += bias[n];
          v = v > 0.f ? v : 0.f;
          Cb[(size_t)m * ldc + n] = (__bf16)v;
        } else if (EPI == EPI_QKV) {   // [B,H,S,64]
          const int bt = m >> 10, s = m & 1023;
          const int h = n >> 6, d = n & 63;
          Cb[((size_t)(bt * 16 + h) * 1024 + s) * 64 + d] = (__bf16)v;
        } else {                        // EPI_VT: [B,H,64,S]
          const int bt = m >> 10, s = m & 1023;
          const int h = n >> 6, d = n & 63;
          Cb[((size_t)(bt * 16 + h) * 64 + d) * 1024 + s] = (__bf16)v;
        }
      }
    }
  }
}

// ---------------------------------------------------------------------------
// Flash attention: grid (B*H=64, S/64=16), block 128 (4 waves),
// wave = 16 query rows, streaming 16 key blocks of 64.
// Q,K: [B,H,S,64] bf16; Vt: [B,H,64,S] bf16; O: [M, D] bf16 (token-major).
// Row sums of P computed with a ones-matrix WMMA (replaces DS shuffles).
// ---------------------------------------------------------------------------
__global__ __launch_bounds__(128)
void attention_kernel(const __bf16* __restrict__ Q, const __bf16* __restrict__ Kk,
                      const __bf16* __restrict__ Vt, __bf16* __restrict__ O) {
  __shared__ __align__(16) __bf16 pbuf[4][16 * 64];
  const int lane  = threadIdx.x & 31;
  const int w     = threadIdx.x >> 5;
  const int bh    = blockIdx.x;
  const int qrow0 = blockIdx.y * 64 + w * 16;

  const __bf16* Qb = Q  + (size_t)bh * 1024 * 64;
  const __bf16* Kb = Kk + (size_t)bh * 1024 * 64;
  const __bf16* Vb = Vt + (size_t)bh * 64 * 1024;

  Frag qf0, qf1;
  load_a(qf0, Qb, 64, qrow0, 0,  lane);
  load_a(qf1, Qb, 64, qrow0, 32, lane);

  Frag ones;
#pragma unroll
  for (int e = 0; e < 16; e++) ones.v[e] = (__bf16)1.0f;

  v8f  oacc[4] = {};
  float mrow[8], lrow[8], alpha[8];
#pragma unroll
  for (int r = 0; r < 8; r++) { mrow[r] = -1e30f; lrow[r] = 0.f; }

  const int nlane  = lane & 15;
  const int rowadd = (lane < 16) ? 0 : 8;

  for (int kb = 0; kb < 16; kb++) {
    const int key0 = kb * 64;
    // S_tile = Q @ K^T  (16 x 64)
    v8f sc[4];
#pragma unroll
    for (int nt = 0; nt < 4; nt++) {
      v8f a = {};
      Frag kf;
      load_b(kf, Kb, 64, key0 + 16 * nt, 0,  lane);
      a = wmma_bf16(qf0, kf, a);
      load_b(kf, Kb, 64, key0 + 16 * nt, 32, lane);
      a = wmma_bf16(qf1, kf, a);
      sc[nt] = a;
    }
    // online softmax: row max via half-wave shuffle butterfly
#pragma unroll
    for (int r = 0; r < 8; r++) {
      float tm = -1e30f;
#pragma unroll
      for (int nt = 0; nt < 4; nt++) tm = fmaxf(tm, sc[nt][r] * 0.125f);
#pragma unroll
      for (int msk = 1; msk < 16; msk <<= 1) tm = fmaxf(tm, __shfl_xor(tm, msk, 32));
      const float nm = fmaxf(mrow[r], tm);
      alpha[r] = __expf(mrow[r] - nm);
      mrow[r]  = nm;
#pragma unroll
      for (int nt = 0; nt < 4; nt++) oacc[nt][r] *= alpha[r];
#pragma unroll
      for (int nt = 0; nt < 4; nt++) sc[nt][r] = __expf(sc[nt][r] * 0.125f - nm);
    }
    // P (C-layout) -> wave-private LDS slab -> A-fragment re-layout
#pragma unroll
    for (int nt = 0; nt < 4; nt++)
#pragma unroll
      for (int r = 0; r < 8; r++)
        pbuf[w][(rowadd + r) * 64 + nt * 16 + nlane] = (__bf16)sc[nt][r];
    asm volatile("s_wait_dscnt 0x0" ::: "memory");   // intra-wave DS ordering

    Frag pf0, pf1;
    load_a(pf0, &pbuf[w][0], 64, 0, 0,  lane);
    load_a(pf1, &pbuf[w][0], 64, 0, 32, lane);

    // row sums of P via ones-matrix WMMA (result replicated across columns)
    v8f rs = {};
    rs = wmma_bf16(pf0, ones, rs);
    rs = wmma_bf16(pf1, ones, rs);
#pragma unroll
    for (int r = 0; r < 8; r++) lrow[r] = lrow[r] * alpha[r] + rs[r];

    // O += P @ V   (V transposed: rows are d, keys contiguous)
#pragma unroll
    for (int nt = 0; nt < 4; nt++) {
      v8f a = oacc[nt];
      Frag vf;
      load_b(vf, Vb, 1024, nt * 16, key0,      lane);
      a = wmma_bf16(pf0, vf, a);
      load_b(vf, Vb, 1024, nt * 16, key0 + 32, lane);
      a = wmma_bf16(pf1, vf, a);
      oacc[nt] = a;
    }
  }

  const int h  = bh & 15;
  const int bt = bh >> 4;
#pragma unroll
  for (int nt = 0; nt < 4; nt++)
#pragma unroll
    for (int r = 0; r < 8; r++) {
      const float v = oacc[nt][r] / lrow[r];
      const int s = qrow0 + rowadd + r;
      const int d = nt * 16 + nlane;
      O[((size_t)(bt * 1024 + s)) * 1024 + h * 64 + d] = (__bf16)v;
    }
}

// ---------------------------------------------------------------------------
// LayerNorm over D=1024, one block per row, dual f32 + bf16 output.
// ---------------------------------------------------------------------------
__global__ __launch_bounds__(256)
void layernorm_kernel(const float* __restrict__ in, const float* __restrict__ g,
                      const float* __restrict__ b, float* __restrict__ out,
                      __bf16* __restrict__ outb) {
  const int row = blockIdx.x;
  const float* x = in + (size_t)row * 1024;
  __shared__ float s1[8], s2[8];
  float v[4], a1 = 0.f, a2 = 0.f;
#pragma unroll
  for (int i = 0; i < 4; i++) {
    v[i] = x[threadIdx.x + 256 * i];
    a1 += v[i];
    a2 += v[i] * v[i];
  }
#pragma unroll
  for (int msk = 16; msk; msk >>= 1) {
    a1 += __shfl_xor(a1, msk, 32);
    a2 += __shfl_xor(a2, msk, 32);
  }
  const int w = threadIdx.x >> 5, lane = threadIdx.x & 31;
  if (lane == 0) { s1[w] = a1; s2[w] = a2; }
  __syncthreads();
  float t1 = 0.f, t2 = 0.f;
#pragma unroll
  for (int i = 0; i < 8; i++) { t1 += s1[i]; t2 += s2[i]; }
  const float mean = t1 * (1.f / 1024.f);
  const float var  = t2 * (1.f / 1024.f) - mean * mean;
  const float rstd = rsqrtf(var + 1e-5f);
#pragma unroll
  for (int i = 0; i < 4; i++) {
    const int c = threadIdx.x + 256 * i;
    const float y = (v[i] - mean) * rstd * g[c] + b[c];
    out[(size_t)row * 1024 + c]  = y;
    outb[(size_t)row * 1024 + c] = (__bf16)y;
  }
}

// ---------------------------------------------------------------------------
// angles = x2 @ ql1_w^T + ql1_b   (M x 8, K=1024) — tiny fp32 GEMM.
// ---------------------------------------------------------------------------
__global__ void angles_kernel(const float* __restrict__ x2,
                              const float* __restrict__ qw,
                              const float* __restrict__ qb,
                              float* __restrict__ ang, int M) {
  const int t = blockIdx.x * blockDim.x + threadIdx.x;
  if (t >= M * 8) return;
  const int m = t >> 3, n = t & 7;
  const float* xr = x2 + (size_t)m * 1024;
  const float* wr = qw + (size_t)n * 1024;
  float acc = qb[n];
#pragma unroll 4
  for (int k = 0; k < 1024; k++) acc += xr[k] * wr[k];
  ang[t] = acc;
}

// ---------------------------------------------------------------------------
// 8-qubit statevector: one wave32 per token. Amplitude index a in [0,256):
// lane = a>>3 (bits 7..3), j = a&7 (bits 2..0). Wire w <-> bit (7-w).
// ---------------------------------------------------------------------------
template <int BIT>
__device__ __forceinline__ void rx_gate(float2 (&s)[8], float c, float sn, int lane) {
  if constexpr (BIT < 3) {
    float2 t[8];
#pragma unroll
    for (int j = 0; j < 8; j++) {
      const float2 p = s[j ^ (1 << BIT)];
      t[j].x = c * s[j].x + sn * p.y;
      t[j].y = c * s[j].y - sn * p.x;
    }
#pragma unroll
    for (int j = 0; j < 8; j++) s[j] = t[j];
  } else {
#pragma unroll
    for (int j = 0; j < 8; j++) {
      const float px = __shfl_xor(s[j].x, 1 << (BIT - 3), 32);
      const float py = __shfl_xor(s[j].y, 1 << (BIT - 3), 32);
      const float nx = c * s[j].x + sn * py;
      const float ny = c * s[j].y - sn * px;
      s[j].x = nx; s[j].y = ny;
    }
  }
}

template <int CBIT, int TBIT>
__device__ __forceinline__ void cnot_gate(float2 (&s)[8], int lane) {
  float2 t[8];
#pragma unroll
  for (int j = 0; j < 8; j++) {
    float2 f;
    if constexpr (TBIT < 3) {
      f = s[j ^ (1 << TBIT)];
    } else {
      f.x = __shfl_xor(s[j].x, 1 << (TBIT - 3), 32);
      f.y = __shfl_xor(s[j].y, 1 << (TBIT - 3), 32);
    }
    bool ctrl;
    if constexpr (CBIT < 3) ctrl = (j >> CBIT) & 1;
    else                    ctrl = (lane >> (CBIT - 3)) & 1;
    t[j] = ctrl ? f : s[j];
  }
#pragma unroll
  for (int j = 0; j < 8; j++) s[j] = t[j];
}

__global__ __launch_bounds__(256)
void quantum_kernel(const float* __restrict__ ang, const float* __restrict__ qw,
                    float* __restrict__ z, int ntok) {
  const int lane = threadIdx.x & 31;
  const int tok  = blockIdx.x * 8 + (threadIdx.x >> 5);
  if (tok >= ntok) return;

  float2 s[8];
#pragma unroll
  for (int j = 0; j < 8; j++) s[j] = make_float2(0.f, 0.f);
  if (lane == 0) s[0].x = 1.f;

  float th[8];
#pragma unroll
  for (int j = 0; j < 8; j++) th[j] = ang[(size_t)tok * 8 + j];
#define RX_W(W) { const float h_ = th[W] * 0.5f; rx_gate<7 - (W)>(s, __cosf(h_), __sinf(h_), lane); }
  RX_W(0) RX_W(1) RX_W(2) RX_W(3) RX_W(4) RX_W(5) RX_W(6) RX_W(7)
#pragma unroll
  for (int j = 0; j < 8; j++) th[j] = qw[j];
  RX_W(0) RX_W(1) RX_W(2) RX_W(3) RX_W(4) RX_W(5) RX_W(6) RX_W(7)
#undef RX_W
  // CNOT ring cw=0..7, tw=(cw+1)%8 ; bits = 7-wire.
  cnot_gate<7, 6>(s, lane);
  cnot_gate<6, 5>(s, lane);
  cnot_gate<5, 4>(s, lane);
  cnot_gate<4, 3>(s, lane);
  cnot_gate<3, 2>(s, lane);
  cnot_gate<2, 1>(s, lane);
  cnot_gate<1, 0>(s, lane);
  cnot_gate<0, 7>(s, lane);

#pragma unroll
  for (int w = 0; w < 8; w++) {
    const int bit = 7 - w;
    float acc = 0.f;
#pragma unroll
    for (int j = 0; j < 8; j++) {
      const float p = s[j].x * s[j].x + s[j].y * s[j].y;
      const int b = (bit < 3) ? ((j >> bit) & 1) : ((lane >> (bit - 3)) & 1);
      acc += b ? -p : p;
    }
#pragma unroll
    for (int msk = 16; msk; msk >>= 1) acc += __shfl_xor(acc, msk, 32);
    if (lane == 0) z[(size_t)tok * 8 + w] = acc;
  }
}

// ---------------------------------------------------------------------------
// y[m,n] += ql2_b[n] + sum_{k<8} z[m,k] * ql2_w[n,k]   (rank-8 fix-up)
// ---------------------------------------------------------------------------
__global__ void finalize_kernel(float* __restrict__ y, const float* __restrict__ z,
                                const float* __restrict__ w2,
                                const float* __restrict__ b2, int M) {
  const int t = blockIdx.x * blockDim.x + threadIdx.x;
  if (t >= M * 1024) return;
  const int m = t >> 10, n = t & 1023;
  float acc = y[t] + b2[n];
  const float* zr = z + (size_t)m * 8;
  const float* wr = w2 + (size_t)n * 1032;
#pragma unroll
  for (int k = 0; k < 8; k++) acc += zr[k] * wr[k];
  y[t] = acc;
}

// ---------------------------------------------------------------------------
// Host launcher
// ---------------------------------------------------------------------------
extern "C" void kernel_launch(void* const* d_in, const int* in_sizes, int n_in,
                              void* d_out, int out_size, void* d_ws, size_t ws_size,
                              hipStream_t stream) {
  const float* x    = (const float*)d_in[0];
  const float* Wq   = (const float*)d_in[1];
  const float* Wk   = (const float*)d_in[2];
  const float* Wv   = (const float*)d_in[3];
  const float* Wo   = (const float*)d_in[4];
  const float* ln1g = (const float*)d_in[5];
  const float* ln1b = (const float*)d_in[6];
  const float* W1   = (const float*)d_in[7];
  const float* b1   = (const float*)d_in[8];
  const float* W2   = (const float*)d_in[9];
  const float* b2   = (const float*)d_in[10];
  const float* ln2g = (const float*)d_in[11];
  const float* ln2b = (const float*)d_in[12];
  const float* ql1w = (const float*)d_in[13];
  const float* ql1b = (const float*)d_in[14];
  const float* qwts = (const float*)d_in[15];
  const float* ql2w = (const float*)d_in[16];
  const float* ql2b = (const float*)d_in[17];

  constexpr int M = 4096, D = 1024, F = 4096;

  char* p = (char*)d_ws;
  auto alloc = [&](size_t bytes) -> char* {
    char* r = p;
    p += (bytes + 255) & ~(size_t)255;
    return r;
  };
  __bf16* xb    = (__bf16*)alloc((size_t)M * D * 2);
  __bf16* Wqb   = (__bf16*)alloc((size_t)D * D * 2);
  __bf16* Wkb   = (__bf16*)alloc((size_t)D * D * 2);
  __bf16* Wvb   = (__bf16*)alloc((size_t)D * D * 2);
  __bf16* Wob   = (__bf16*)alloc((size_t)D * D * 2);
  __bf16* W1b   = (__bf16*)alloc((size_t)F * D * 2);
  __bf16* W2b   = (__bf16*)alloc((size_t)D * F * 2);
  __bf16* ql2wb = (__bf16*)alloc((size_t)D * 1032 * 2);
  __bf16* qb    = (__bf16*)alloc((size_t)M * D * 2);
  __bf16* kb    = (__bf16*)alloc((size_t)M * D * 2);
  __bf16* vtb   = (__bf16*)alloc((size_t)M * D * 2);
  __bf16* ob    = (__bf16*)alloc((size_t)M * D * 2);
  float*  h1    = (float*)alloc((size_t)M * D * 4);
  float*  x1    = (float*)alloc((size_t)M * D * 4);
  __bf16* x1b   = (__bf16*)alloc((size_t)M * D * 2);
  __bf16* ff1b  = (__bf16*)alloc((size_t)M * F * 2);
  float*  h2    = (float*)alloc((size_t)M * D * 4);
  float*  x2    = (float*)alloc((size_t)M * D * 4);
  __bf16* x2b   = (__bf16*)alloc((size_t)M * D * 2);
  float*  ang   = (float*)alloc((size_t)M * 8 * 4);
  float*  zbuf  = (float*)alloc((size_t)M * 8 * 4);

  auto cvt = [&](const float* in, __bf16* out, long n) {
    cvt_bf16_kernel<<<1024, 256, 0, stream>>>(in, out, n / 8);
  };
  cvt(x,    xb,    (long)M * D);
  cvt(Wq,   Wqb,   (long)D * D);
  cvt(Wk,   Wkb,   (long)D * D);
  cvt(Wv,   Wvb,   (long)D * D);
  cvt(Wo,   Wob,   (long)D * D);
  cvt(W1,   W1b,   (long)F * D);
  cvt(W2,   W2b,   (long)D * F);
  cvt(ql2w, ql2wb, (long)D * 1032);

  const dim3 gD(D / 256, M / 64);   // N=1024 GEMMs
  const dim3 gF(F / 256, M / 64);   // N=4096 GEMM

  // QKV projections with scatter epilogues
  gemm_bf16_kernel<EPI_QKV><<<gD, 256, 0, stream>>>(xb, Wqb, D, D, D, nullptr, nullptr, nullptr, qb, D);
  gemm_bf16_kernel<EPI_QKV><<<gD, 256, 0, stream>>>(xb, Wkb, D, D, D, nullptr, nullptr, nullptr, kb, D);
  gemm_bf16_kernel<EPI_VT> <<<gD, 256, 0, stream>>>(xb, Wvb, D, D, D, nullptr, nullptr, nullptr, vtb, D);

  // Flash attention
  attention_kernel<<<dim3(64, 16), 128, 0, stream>>>(qb, kb, vtb, ob);

  // h1 = o @ Wo^T + x ; LN1
  gemm_bf16_kernel<EPI_F32><<<gD, 256, 0, stream>>>(ob, Wob, D, D, D, nullptr, x, h1, nullptr, D);
  layernorm_kernel<<<M, 256, 0, stream>>>(h1, ln1g, ln1b, x1, x1b);

  // FFN
  gemm_bf16_kernel<EPI_RELU_BF16><<<gF, 256, 0, stream>>>(x1b, W1b, D, D, D, b1, nullptr, nullptr, ff1b, F);
  gemm_bf16_kernel<EPI_F32><<<gD, 256, 0, stream>>>(ff1b, W2b, F, F, F, b2, x1, h2, nullptr, D);
  layernorm_kernel<<<M, 256, 0, stream>>>(h2, ln2g, ln2b, x2, x2b);

  // quantum head
  angles_kernel<<<(M * 8 + 255) / 256, 256, 0, stream>>>(x2, ql1w, ql1b, ang, M);
  quantum_kernel<<<M / 8, 256, 0, stream>>>(ang, qwts, zbuf, M);

  // y = x2 @ ql2_w[:,8:]^T  (main K=1024 WMMA GEMM, straight into d_out)
  gemm_bf16_kernel<EPI_F32><<<gD, 256, 0, stream>>>(x2b, ql2wb + 8, D, D, 1032,
                                                    nullptr, nullptr, (float*)d_out, nullptr, D);
  // y += bias + z @ ql2_w[:,:8]^T
  finalize_kernel<<<(M * 1024 + 255) / 256, 256, 0, stream>>>((float*)d_out, zbuf, ql2w, ql2b, M);
}
